// LocationSensitiveAttention_40407052320846
// MI455X (gfx1250) — matrix operand order repaired and checked by reading, not compiled, not run on hardware
//
#include <hip/hip_runtime.h>
#include <hip/hip_bf16.h>

// Location-sensitive attention, fused for MI455X (gfx1250, wave32).
// - Main GEMM: v_wmma_f32_16x16x32_bf16, K = 512 (memory) + 32 (loc features)
// - processed_query folded into WMMA accumulator init
// - conv1d location features staged in LDS as the 17th K-chunk
// - memory tiles staged via TENSOR_LOAD_TO_LDS (TDM), double-buffered,
//   synchronized with s_wait_tensorcnt + workgroup barrier
// Memory-bound: 2 passes over `memory` (2 x 256 MB) ~ 22 us at 23.3 TB/s.

typedef __attribute__((ext_vector_type(16))) __bf16 v16bf;
typedef __attribute__((ext_vector_type(8)))  float  v8f;
typedef __attribute__((ext_vector_type(4)))  unsigned int u32x4;
typedef __attribute__((ext_vector_type(8)))  int          i32x8;
typedef __attribute__((ext_vector_type(4)))  int          i32x4;

#define Bdim   64
#define Tdim   2048
#define Edim   512
#define Adim   128
#define RNNdim 1024
#define Fdim   32
#define KS     31
#define KCAT   544      // 512 + 32
#define TT     128      // t-rows per workgroup
#define RS_H   36       // sLoc row stride in halfs (72B, conflict-free)
#define RS_W   18       // sLoc row stride in dwords
#define FRS    34       // f32 tile row stride in dwords (128B + 8B TDM pad)

__device__ __forceinline__ unsigned short f32_to_bf16_bits(float f) {
  unsigned int u = __float_as_uint(f);
  u += 0x7FFFu + ((u >> 16) & 1u);   // round-to-nearest-even
  return (unsigned short)(u >> 16);
}
__device__ __forceinline__ unsigned int pack_bf16x2(float a, float b) {
  return (unsigned int)f32_to_bf16_bits(a) |
         ((unsigned int)f32_to_bf16_bits(b) << 16);
}

// ---- TDM: DMA a 128x32 f32 tile (row stride 512 f32) into LDS ----
// D# per CDNA5 ISA. LDS pad: 2 dwords per 32 dwords
// -> LDS row stride 136 B (= FRS dwords).
__device__ __forceinline__ void tdm_load_tile(const float* gsrc, unsigned lds_addr) {
  unsigned long long ga = (unsigned long long)(uintptr_t)gsrc;
  u32x4 g0;
  g0[0] = 1u;                                        // count=1, user mode
  g0[1] = lds_addr;                                  // LDS byte offset
  g0[2] = (unsigned int)(ga & 0xFFFFFFFFull);        // global_addr lo
  g0[3] = (unsigned int)((ga >> 32) & 0x01FFFFFFull) // global_addr hi
          | 0x80000000u;                             // type=2 ("image")
  i32x8 g1;
  g1[0] = (int)((2u << 16)      // data_size = 4B
              | (1u << 20)      // pad_enable
              | (4u << 22)      // pad_interval: 32 dwords
              | (1u << 25));    // pad_amount: 2 dwords
  g1[1] = (int)(512u << 16);    // tensor_dim0 = 512
  g1[2] = (int)(2048u << 16);   // tensor_dim1 = 2048 (lo16 in [31:16])
  g1[3] = (int)(32u << 16);     // tile_dim0 = 32
  g1[4] = (int)(128u);          // tile_dim1 = 128, tile_dim2 = 0 (unused)
  g1[5] = (int)(512u);          // tensor_dim0_stride = 512
  g1[6] = 0;                    // stride0 hi / stride1 lo
  g1[7] = 0;                    // stride1 hi
  i32x4 g2; g2[0] = 1; g2[1] = 1; g2[2] = 0; g2[3] = 0;        // dim2=dim3=1
  i32x4 g3; g3[0] = 0; g3[1] = (int)(1u << 16); g3[2] = 0; g3[3] = 0; // dim4=1
  i32x8 g4;                                                     // unused group
  g4[0] = 0; g4[1] = 0; g4[2] = 0; g4[3] = 0;
  g4[4] = 0; g4[5] = 0; g4[6] = 0; g4[7] = 0;
  __builtin_amdgcn_tensor_load_to_lds(g0, g1, g2, g3, g4, 0);
}

// ---- prep: Wcat[a][0..511]=W_memory[a][e], [512..543]=W_loc[a][f], bf16 ----
__global__ void prep_wcat_kernel(const float* __restrict__ Wm,
                                 const float* __restrict__ Wl,
                                 unsigned short* __restrict__ wcat) {
  int i = blockIdx.x * 256 + threadIdx.x;
  if (i >= Adim * KCAT) return;
  int a = i / KCAT, j = i % KCAT;
  float x = (j < Edim) ? Wm[a * Edim + j] : Wl[a * Fdim + (j - Edim)];
  wcat[i] = f32_to_bf16_bits(x);
}

// ---- prep: pq[b][a] = h[b] . W_query[a] ----
__global__ void prep_pq_kernel(const float* __restrict__ h,
                               const float* __restrict__ Wq,
                               float* __restrict__ pq) {
  int b = blockIdx.x, a = threadIdx.x;
  const float4* hr = (const float4*)(h + b * RNNdim);
  const float4* wr = (const float4*)(Wq + a * RNNdim);
  float acc = 0.f;
  for (int r = 0; r < RNNdim / 4; ++r) {
    float4 x = hr[r], y = wr[r];
    acc += x.x * y.x + x.y * y.y + x.z * y.z + x.w * y.w;
  }
  pq[b * Adim + a] = acc;
}

// ---- main: fused conv + GEMM (WMMA bf16) + tanh/v-dot -> energies(B,T) ----
__global__ __launch_bounds__(256) void energies_kernel(
    const float* __restrict__ memory,       // (B,T,E)
    const float* __restrict__ attw,         // (B,2,T)
    const float* __restrict__ conv_w,       // (F,2,KS)
    const unsigned short* __restrict__ wcat,// (A,KCAT) bf16
    const float* __restrict__ pq,           // (B,A)
    const float* __restrict__ v,            // (A)
    float* __restrict__ energy)             // (B,T)
{
  __shared__ __align__(16) float bufA[TT * FRS];          // TDM dest, chunk even
  __shared__ __align__(16) float bufB[TT * FRS];          // TDM dest, chunk odd
  __shared__ __align__(16) unsigned short sLoc[TT * RS_H];

  const int tid  = threadIdx.x;
  const int b    = blockIdx.y;
  const int t0   = blockIdx.x * TT;
  const int wave = tid >> 5;
  const int lane = tid & 31;
  const int lg   = lane & 15;         // lane-in-group (N index / M row)
  const int hi   = (lane >> 4) & 1;   // 0: lanes 0-15, 1: lanes 16-31

  const float* memb = memory + ((size_t)b * Tdim + t0) * Edim;
  const unsigned ldsA = (unsigned)(uintptr_t)bufA;  // low 32 bits = LDS offset
  const unsigned ldsB = (unsigned)(uintptr_t)bufB;

  // kick off DMA of chunk 0 while we compute conv features
  if (wave == 0) tdm_load_tile(memb, ldsA);

  // ---- conv1d location features -> sLoc (bf16, A-frag layout, k=f) ----
  {
    const float* aw = attw + (size_t)b * 2 * Tdim;
    #pragma unroll 4
    for (int rep = 0; rep < 16; ++rep) {
      int idx = rep * 256 + tid;   // 0..4095
      int row = idx >> 5;          // 0..127
      int f   = idx & 31;
      int t   = t0 + row;
      const float* cw = conv_w + f * (2 * KS);
      float acc = 0.f;
      #pragma unroll
      for (int c = 0; c < 2; ++c) {
        for (int j = 0; j < KS; ++j) {
          int tt = t + j - 15;
          float x = (tt >= 0 && tt < Tdim) ? aw[c * Tdim + tt] : 0.f;
          acc += x * cw[c * KS + j];
        }
      }
      sLoc[row * RS_H + f] = f32_to_bf16_bits(acc);
    }
  }

  // ---- accumulators initialized with processed_query (broadcast per N) ----
  v8f acc[8];
  #pragma unroll
  for (int nt = 0; nt < 8; ++nt) {
    float q = pq[b * Adim + nt * 16 + lg];
    #pragma unroll
    for (int r = 0; r < 8; ++r) acc[nt][r] = q;
  }

  // ---- K loop over 16 memory chunks (TDM double-buffered) ----
  for (int kk = 0; kk < 16; ++kk) {
    __builtin_amdgcn_s_wait_tensorcnt(0);   // chunk kk landed (wave0's counter)
    __syncthreads();                        // visible to all; prev compute done
    if (wave == 0 && kk + 1 < 16)
      tdm_load_tile(memb + (kk + 1) * 32, (kk & 1) ? ldsA : ldsB);

    // A fragment: read f32 pairs from LDS tile, pack to bf16
    const float* fb = (kk & 1) ? bufB : bufA;
    int rbase = (wave * 16 + lg) * FRS + hi * 8;
    union { unsigned int u[8]; v16bf v; } af;
    #pragma unroll
    for (int d = 0; d < 4; ++d) {
      float2 p = *(const float2*)(fb + rbase + 2 * d);
      float2 q = *(const float2*)(fb + rbase + 16 + 2 * d);
      af.u[d]     = pack_bf16x2(p.x, p.y);
      af.u[d + 4] = pack_bf16x2(q.x, q.y);
    }

    int col0 = kk * 32;
    #pragma unroll
    for (int nt = 0; nt < 8; ++nt) {
      union { uint4 q[2]; v16bf v; } bf;
      const uint4* bp = (const uint4*)(wcat +
          (size_t)(nt * 16 + lg) * KCAT + col0 + hi * 8);
      bf.q[0] = bp[0];
      bf.q[1] = bp[2];
      acc[nt] = __builtin_amdgcn_wmma_f32_16x16x32_bf16(
          false, af.v, false, bf.v, (short)0, acc[nt], false, false);
    }
  }

  // ---- 17th chunk: location features (bf16 already in sLoc) ----
  {
    const unsigned int* abase = (const unsigned int*)sLoc;
    union { unsigned int u[8]; v16bf v; } af;
    int base = (wave * 16 + lg) * RS_W + hi * 4;
    #pragma unroll
    for (int d = 0; d < 4; ++d) {
      af.u[d]     = abase[base + d];
      af.u[d + 4] = abase[base + 8 + d];
    }
    #pragma unroll
    for (int nt = 0; nt < 8; ++nt) {
      union { uint4 q[2]; v16bf v; } bf;
      const uint4* bp = (const uint4*)(wcat +
          (size_t)(nt * 16 + lg) * KCAT + 512 + hi * 8);
      bf.q[0] = bp[0];
      bf.q[1] = bp[2];
      acc[nt] = __builtin_amdgcn_wmma_f32_16x16x32_bf16(
          false, af.v, false, bf.v, (short)0, acc[nt], false, false);
    }
  }

  // ---- epilogue: energy[t] = sum_a v[a]*tanh(D[t][a]) ----
  float e[8];
  #pragma unroll
  for (int r = 0; r < 8; ++r) e[r] = 0.f;
  #pragma unroll
  for (int nt = 0; nt < 8; ++nt) {
    float va = v[nt * 16 + lg];
    #pragma unroll
    for (int r = 0; r < 8; ++r) e[r] += va * tanhf(acc[nt][r]);
  }
  #pragma unroll
  for (int r = 0; r < 8; ++r) {        // reduce over 16 N-lanes (stays in group)
    float x = e[r];
    x += __shfl_xor(x, 1, 32);
    x += __shfl_xor(x, 2, 32);
    x += __shfl_xor(x, 4, 32);
    x += __shfl_xor(x, 8, 32);
    e[r] = x;
  }
  if (lg == 0) {
    #pragma unroll
    for (int r = 0; r < 8; ++r) {
      int m = hi * 8 + r;              // lanes 0-15: M=0..7, 16-31: M=8..15
      energy[(size_t)b * Tdim + t0 + wave * 16 + m] = e[r];
    }
  }
}

// ---- softmax over T per batch ----
__global__ __launch_bounds__(256) void softmax_kernel(
    const float* __restrict__ energy,
    const unsigned char* __restrict__ mask,
    float* __restrict__ weights)
{
  __shared__ float red[8];
  int b = blockIdx.x, tid = threadIdx.x;
  int lane = tid & 31, wave = tid >> 5;
  float vals[8];
  float mx = -1e30f;
  #pragma unroll
  for (int i = 0; i < 8; ++i) {
    int t = i * 256 + tid;
    float x = energy[(size_t)b * Tdim + t];
    if (mask[(size_t)b * Tdim + t]) x = -1e30f;
    vals[i] = x;
    mx = fmaxf(mx, x);
  }
  for (int off = 16; off; off >>= 1) mx = fmaxf(mx, __shfl_xor(mx, off, 32));
  if (lane == 0) red[wave] = mx;
  __syncthreads();
  float gmx = red[0];
  #pragma unroll
  for (int w = 1; w < 8; ++w) gmx = fmaxf(gmx, red[w]);
  __syncthreads();

  float s = 0.f;
  #pragma unroll
  for (int i = 0; i < 8; ++i) { vals[i] = __expf(vals[i] - gmx); s += vals[i]; }
  for (int off = 16; off; off >>= 1) s += __shfl_xor(s, off, 32);
  if (lane == 0) red[wave] = s;
  __syncthreads();
  float tot = 0.f;
  #pragma unroll
  for (int w = 0; w < 8; ++w) tot += red[w];
  float inv = 1.f / tot;
  #pragma unroll
  for (int i = 0; i < 8; ++i)
    weights[(size_t)b * Tdim + i * 256 + tid] = vals[i] * inv;
}

// ---- context: ctx[b][e] = sum_t w[b][t] * memory[b][t][e], float4 wide ----
__global__ __launch_bounds__(256) void context_kernel(
    const float* __restrict__ memory,
    const float* __restrict__ weights,
    float* __restrict__ ctx)
{
  __shared__ float4 red[256];
  int b = blockIdx.x;
  int lq    = threadIdx.x & 31;    // 32 e-quads = 128 e per block
  int slice = threadIdx.x >> 5;    // 8 t-slices
  int e0 = blockIdx.y * 128 + lq * 4;
  const float*  w  = weights + (size_t)b * Tdim;
  const float4* mb = (const float4*)(memory + (size_t)b * Tdim * Edim);
  float4 acc = make_float4(0.f, 0.f, 0.f, 0.f);
  for (int t = slice; t < Tdim; t += 8) {
    float wt = w[t];
    float4 m = mb[(size_t)t * (Edim / 4) + (e0 >> 2)];
    acc.x += wt * m.x; acc.y += wt * m.y;
    acc.z += wt * m.z; acc.w += wt * m.w;
  }
  red[threadIdx.x] = acc;
  __syncthreads();
  if (slice == 0) {
    #pragma unroll
    for (int s = 1; s < 8; ++s) {
      float4 r = red[lq + s * 32];
      acc.x += r.x; acc.y += r.y; acc.z += r.z; acc.w += r.w;
    }
    *(float4*)&ctx[(size_t)b * Edim + e0] = acc;
  }
}

extern "C" void kernel_launch(void* const* d_in, const int* in_sizes, int n_in,
                              void* d_out, int out_size, void* d_ws, size_t ws_size,
                              hipStream_t stream) {
  (void)in_sizes; (void)n_in; (void)out_size; (void)ws_size;
  const float* h      = (const float*)d_in[0];        // (B,RNN)
  const float* memory = (const float*)d_in[1];        // (B,T,E)
  const float* attw   = (const float*)d_in[2];        // (B,2,T)
  const unsigned char* mask = (const unsigned char*)d_in[3]; // (B,T) bool
  const float* Wq     = (const float*)d_in[4];        // (A,RNN)
  const float* Wm     = (const float*)d_in[5];        // (A,E)
  const float* v      = (const float*)d_in[6];        // (1,A)
  const float* cw     = (const float*)d_in[7];        // (F,2,KS)
  const float* Wl     = (const float*)d_in[8];        // (A,F)

  // workspace layout
  char* ws = (char*)d_ws;
  unsigned short* wcat = (unsigned short*)(ws);                    // 139264 B
  float* pq     = (float*)(ws + 139264);                           // 32768 B
  float* energy = (float*)(ws + 139264 + 32768);                   // 524288 B

  float* out     = (float*)d_out;
  float* ctx     = out;                       // (B,1,E) = 32768 floats
  float* weights = out + Bdim * Edim;         // (B,T)   = 131072 floats

  prep_wcat_kernel<<<(Adim * KCAT + 255) / 256, 256, 0, stream>>>(Wm, Wl, wcat);
  prep_pq_kernel<<<Bdim, Adim, 0, stream>>>(h, Wq, pq);
  energies_kernel<<<dim3(Tdim / TT, Bdim), 256, 0, stream>>>(
      memory, attw, cw, wcat, pq, v, energy);
  softmax_kernel<<<Bdim, 256, 0, stream>>>(energy, mask, weights);
  context_kernel<<<dim3(Bdim, Edim / 64), 256, 0, stream>>>(memory, weights, ctx);
}
